// Model_24292335026625
// MI455X (gfx1250) — matrix operand order repaired and checked by reading, not compiled, not run on hardware
//
#include <hip/hip_runtime.h>
#include <hip/hip_bf16.h>
#include <math.h>

#define B_   128
#define L_   1024
#define H_   256
#define NC_  32
#define NB_  6

typedef __attribute__((ext_vector_type(16))) __bf16 v16bf;
typedef __attribute__((ext_vector_type(8)))  float  v8f;
typedef __attribute__((ext_vector_type(4)))  unsigned int u32x4;
typedef __attribute__((ext_vector_type(4)))  int i32x4;
typedef __attribute__((ext_vector_type(8)))  int i32x8;

struct alignas(16) U4 { unsigned int x, y, z, w; };

union FragBF {
    v16bf v;
    unsigned short us[16];
    U4 q[2];
};

__device__ __forceinline__ unsigned short f2bf(float f) {
    unsigned int u = __float_as_uint(f);
    u += 0x7FFFu + ((u >> 16) & 1u);   // round-to-nearest-even
    return (unsigned short)(u >> 16);
}

// ---------------------------------------------------------------------------
// 1) r[b,l,h] = LN_h( x[b,l]*pre_w[h] + pre_b[h] ) * emb_g + emb_b
// ---------------------------------------------------------------------------
__global__ void pre_ln_kernel(const float* __restrict__ x, const float* __restrict__ pre_w,
                              const float* __restrict__ pre_b, const float* __restrict__ emb_g,
                              const float* __restrict__ emb_b, float* __restrict__ r) {
    int row = blockIdx.x;            // b*L + l
    int h = threadIdx.x;
    float v = x[row] * pre_w[h] + pre_b[h];
    __shared__ float red[8];
    int lane = h & 31, wid = h >> 5;
    float s = v;
    for (int m = 16; m; m >>= 1) s += __shfl_xor(s, m, 32);
    if (lane == 0) red[wid] = s;
    __syncthreads();
    float mu = 0.f;
    for (int i = 0; i < 8; ++i) mu += red[i];
    mu *= (1.0f / 256.0f);
    __syncthreads();
    float d = v - mu;
    float s2 = d * d;
    for (int m = 16; m; m >>= 1) s2 += __shfl_xor(s2, m, 32);
    if (lane == 0) red[wid] = s2;
    __syncthreads();
    float var = 0.f;
    for (int i = 0; i < 8; ++i) var += red[i];
    var *= (1.0f / 256.0f);
    float rs = rsqrtf(var + 1e-5f);
    r[(size_t)row * H_ + h] = d * rs * emb_g[h] + emb_b[h];
}

// ---------------------------------------------------------------------------
// 2) per-block discretization: lambda = exp(dt*A), Ct = 2 * C*(lambda-1)/A
// ---------------------------------------------------------------------------
__global__ void disc_kernel(const float* __restrict__ log_dt, const float* __restrict__ A_re,
                            const float* __restrict__ A_im, const float* __restrict__ C_re,
                            const float* __restrict__ C_im, float* __restrict__ lam_r,
                            float* __restrict__ lam_i, float* __restrict__ ct_r,
                            float* __restrict__ ct_i) {
    int t = blockIdx.x * 256 + threadIdx.x;          // H*NC = 8192
    if (t >= H_ * NC_) return;
    int h = t >> 5;
    float dt = __expf(log_dt[h]);
    float ar = A_re[t], ai = A_im[t];
    float cr = C_re[t], ci = C_im[t];
    float dr = ar * dt, di = ai * dt;
    float e  = __expf(dr);
    float lr = e * __cosf(di), li = e * __sinf(di);
    float nr = lr - 1.0f, ni = li;                   // exp(dtA)-1
    float inv = 1.0f / (ar * ar + ai * ai);          // |A|^2 (never 0: A_re=-0.5)
    float tr = (nr * ar + ni * ai) * inv;            // (lambda-1)/A
    float ti = (ni * ar - nr * ai) * inv;
    lam_r[t] = lr; lam_i[t] = li;
    ct_r[t] = 2.0f * (cr * tr - ci * ti);            // fold factor 2
    ct_i[t] = 2.0f * (cr * ti + ci * tr);
}

// Wo (H,H) f32, row n col k  ->  TRANSPOSED bf16: wobT[k*H + n] = bf16(Wo[n,k])
// gives B[k,n] contiguous in n; the WMMA B-fragment is 16 separate 2-byte LDS loads (stride H).
__global__ void wob_kernel(const float* __restrict__ Wo, unsigned short* __restrict__ wobT) {
    int t = blockIdx.x * 256 + threadIdx.x;   // over H*H
    int n = t >> 8;
    int k = t & 255;
    wobT[k * H_ + n] = f2bf(Wo[t]);
}

// ---------------------------------------------------------------------------
// 3) diagonal-SSM scan: one wave32 per (b,h), lane = complex mode n.
//    s = lambda*s + u ;  y = tanh( Re(sum_n Ct*s) + Dp*u )  -> bf16 abuf
// ---------------------------------------------------------------------------
__global__ __launch_bounds__(256) void scan_kernel(
        const float* __restrict__ r, const float* __restrict__ lam_r,
        const float* __restrict__ lam_i, const float* __restrict__ ct_r,
        const float* __restrict__ ct_i, const float* __restrict__ Dp,
        unsigned short* __restrict__ abuf) {
    int wave = (blockIdx.x * blockDim.x + threadIdx.x) >> 5;   // b*H + h
    int lane = threadIdx.x & 31;
    int b = wave / H_;
    int h = wave - b * H_;
    int pidx = h * NC_ + lane;
    float lr = lam_r[pidx], li = lam_i[pidx];
    float cr = ct_r[pidx],  ci = ct_i[pidx];
    float dp = Dp[h];
    float sr = 0.f, si = 0.f;
    const float* ub = r + (size_t)b * L_ * H_ + h;
    unsigned short* ob = abuf + (size_t)b * L_ * H_ + h;
    for (int l0 = 0; l0 < L_; l0 += 32) {
        float ureg = ub[(size_t)(l0 + lane) * H_];   // 32 timesteps staged in regs
        float ysave = 0.f;
        #pragma unroll
        for (int k = 0; k < 32; ++k) {
            float u = __shfl(ureg, k, 32);           // broadcast step-k input
            float nsr = lr * sr - li * si + u;       // complex state update
            float nsi = lr * si + li * sr;
            sr = nsr; si = nsi;
            float p = cr * sr - ci * si;             // Re(Ct * s), per mode
            p += __shfl_xor(p, 16, 32);              // reduce 32 modes
            p += __shfl_xor(p, 8, 32);
            p += __shfl_xor(p, 4, 32);
            p += __shfl_xor(p, 2, 32);
            p += __shfl_xor(p, 1, 32);
            if (lane == k) ysave = p;                // keep own timestep
        }
        float yo = tanhf(ysave + dp * ureg);
        ob[(size_t)(l0 + lane) * H_] = f2bf(yo);
    }
}

// ---------------------------------------------------------------------------
// 4) y2 = y1 @ Wo^T + bo ; r = LN(y2 + r)    (fused epilogue)
//    workgroup: M=128 x N=256, 8 waves (wave w -> cols 16w..16w+15),
//    8 f32 accumulators/wave, K sliced by 32.  Wo^T slice (16KB, contiguous)
//    is staged into LDS by the Tensor Data Mover (tensor_load_to_lds +
//    s_wait_tensorcnt) when available, else by a plain vector copy.
// ---------------------------------------------------------------------------
__global__ __launch_bounds__(256) void gemm_ln_kernel(
        const unsigned short* __restrict__ abuf, const unsigned short* __restrict__ wobT,
        const float* __restrict__ bo, float* __restrict__ rbuf,
        const float* __restrict__ ln_g, const float* __restrict__ ln_b) {
    __shared__ unsigned short sB[32 * H_];   // [k][n] 32x256 bf16 slice (16KB)
    __shared__ float sC[16 * H_];            // 16x256 f32 epilogue tile (16KB)
    int t = threadIdx.x;
    int wid = t >> 5, lane = t & 31;
    int nbase = wid * 16;
    int m0 = blockIdx.x * 128;

    v8f zero = {0.f, 0.f, 0.f, 0.f, 0.f, 0.f, 0.f, 0.f};
    v8f acc[8];
    #pragma unroll
    for (int i = 0; i < 8; ++i) acc[i] = zero;

    int mrow = lane & 15;            // A-matrix: lane -> row (ISA 16-bit A layout)
    int koff = (lane >> 4) * 8;      // lanes 16-31 hold the +8 K chunk

    for (int k0 = 0; k0 < H_; k0 += 32) {
        // ---- stage Wo^T slice rows k0..k0+31 (contiguous 16KB) into sB ----
#if defined(__HIP_DEVICE_COMPILE__) && __has_builtin(__builtin_amdgcn_tensor_load_to_lds) && __has_builtin(__builtin_amdgcn_s_wait_tensorcnt)
        if (wid == 0) {
            // D# group0: count=1 | lds_addr | global_addr | type=2
            unsigned long long ga =
                (unsigned long long)(uintptr_t)wobT + (unsigned long long)k0 * (H_ * 2ull);
            unsigned lds_off = (unsigned)(uintptr_t)(void*)sB;   // low 32b = LDS offset
            u32x4 g0;
            g0.x = 1u;                                           // count=1, user mode
            g0.y = lds_off;
            g0.z = (unsigned)(ga & 0xffffffffull);
            g0.w = (unsigned)((ga >> 32) & 0x1ffffffull) | (2u << 30);  // type=2
            // D# group1: data_size=2B; 1-D tile: tensor_dim0=tile_dim0=8192 elems
            i32x8 g1;
            g1[0] = (int)(1u << 16);          // workgroup_mask=0, data_size=1 (2B)
            g1[1] = (int)(8192u << 16);       // tensor_dim0[15:0] in bits 63:48
            g1[2] = (int)(1u << 16);          // tensor_dim0 hi=0, tensor_dim1=1
            g1[3] = (int)(8192u << 16);       // tile_dim0=8192
            g1[4] = 0;                        // tile_dim1=0 (unused), tile_dim2=0
            g1[5] = 8192;                     // tensor_dim0_stride lo32
            g1[6] = 0;
            g1[7] = 0;
            i32x4 gz = {0, 0, 0, 0};
#if __clang_major__ >= 23
            i32x8 gz8 = {0, 0, 0, 0, 0, 0, 0, 0};
            __builtin_amdgcn_tensor_load_to_lds(g0, g1, gz, gz, gz8, 0);
#else
            __builtin_amdgcn_tensor_load_to_lds(g0, g1, gz, gz, 0);
#endif
        }
        __builtin_amdgcn_s_wait_tensorcnt(0);
#else
        {   // fallback: contiguous vector copy (64B per thread)
            const U4* src = (const U4*)(wobT + (size_t)k0 * H_ + t * 32);
            U4* dst = (U4*)(sB + t * 32);
            dst[0] = src[0]; dst[1] = src[1]; dst[2] = src[2]; dst[3] = src[3];
        }
#endif
        __syncthreads();

        // B frag: lane -> N=nbase+lane%16, element s -> K=koff+s (s<8), 16+koff+s-8 (s>=8)
        FragBF bf;
        {
            const unsigned short* bp = sB + nbase + mrow;
            #pragma unroll
            for (int s = 0; s < 8; ++s) {
                bf.us[s]     = bp[(koff + s) * H_];
                bf.us[s + 8] = bp[(16 + koff + s) * H_];
            }
        }

        #pragma unroll
        for (int ms = 0; ms < 8; ++ms) {
            int rowg = m0 + ms * 16 + mrow;
            const unsigned short* ap = abuf + (size_t)rowg * H_ + k0 + koff;
            FragBF af;
            af.q[0] = *(const U4*)(ap);        // K = k0+koff    .. +7
            af.q[1] = *(const U4*)(ap + 16);   // K = k0+16+koff .. +7
            if (ms == 0 && k0 + 32 < H_)       // global_prefetch_b8 on next A slice
                __builtin_prefetch(ap + 32, 0, 0);
            acc[ms] = __builtin_amdgcn_wmma_f32_16x16x32_bf16(
                          false, af.v, false, bf.v, (short)0, acc[ms], false, false);
        }
        __syncthreads();
    }

    // epilogue: +bo +residual, LayerNorm over H, write back r
    int cm = lane >> 4;              // C layout: VGPR r -> M = r + 8*cm
    int cn = nbase + (lane & 15);    // N = nbase + lane%16
    for (int ms = 0; ms < 8; ++ms) {
        int rbase = m0 + ms * 16;
        #pragma unroll
        for (int rr = 0; rr < 8; ++rr) {
            int M = rr + 8 * cm;
            sC[M * H_ + cn] = acc[ms][rr] + bo[cn] + rbuf[(size_t)(rbase + M) * H_ + cn];
        }
        __syncthreads();
        int row = 2 * wid + cm;      // wave handles 2 rows, 16 lanes each
        int l16 = lane & 15;
        float sum = 0.f;
        #pragma unroll
        for (int i = 0; i < 16; ++i) sum += sC[row * H_ + l16 + 16 * i];
        sum += __shfl_xor(sum, 8, 32);
        sum += __shfl_xor(sum, 4, 32);
        sum += __shfl_xor(sum, 2, 32);
        sum += __shfl_xor(sum, 1, 32);
        float mu = sum * (1.0f / 256.0f);
        float vs = 0.f;
        #pragma unroll
        for (int i = 0; i < 16; ++i) {
            float d = sC[row * H_ + l16 + 16 * i] - mu;
            vs += d * d;
        }
        vs += __shfl_xor(vs, 8, 32);
        vs += __shfl_xor(vs, 4, 32);
        vs += __shfl_xor(vs, 2, 32);
        vs += __shfl_xor(vs, 1, 32);
        float rs = rsqrtf(vs * (1.0f / 256.0f) + 1e-5f);
        int rowg = rbase + row;
        #pragma unroll
        for (int i = 0; i < 16; ++i) {
            int n2 = l16 + 16 * i;
            float d = sC[row * H_ + n2] - mu;
            rbuf[(size_t)rowg * H_ + n2] = d * rs * ln_g[n2] + ln_b[n2];
        }
        __syncthreads();
    }
}

// ---------------------------------------------------------------------------
// 5) head: mean over L, then (B,H) @ fc_w^T + fc_b -> (B,10)
// ---------------------------------------------------------------------------
__global__ void head_kernel(const float* __restrict__ r, const float* __restrict__ fc_w,
                            const float* __restrict__ fc_b, float* __restrict__ out) {
    int b = blockIdx.x;
    int h = threadIdx.x;
    __shared__ float mean[H_];
    float s = 0.f;
    const float* p = r + (size_t)b * L_ * H_ + h;
    for (int l = 0; l < L_; ++l) s += p[(size_t)l * H_];
    mean[h] = s * (1.0f / (1024.0f + 1e-8f));
    __syncthreads();
    if (h < 10) {
        float a = fc_b[h];
        for (int k = 0; k < H_; ++k) a += mean[k] * fc_w[h * H_ + k];
        out[b * 10 + h] = a;
    }
}

// ---------------------------------------------------------------------------
extern "C" void kernel_launch(void* const* d_in, const int* in_sizes, int n_in,
                              void* d_out, int out_size, void* d_ws, size_t ws_size,
                              hipStream_t stream) {
    const float* x      = (const float*)d_in[0];
    const float* pre_w  = (const float*)d_in[1];
    const float* pre_b  = (const float*)d_in[2];
    const float* emb_g  = (const float*)d_in[3];
    const float* emb_b  = (const float*)d_in[4];
    const float* log_dt = (const float*)d_in[5];
    const float* A_re   = (const float*)d_in[6];
    const float* A_im   = (const float*)d_in[7];
    const float* C_re   = (const float*)d_in[8];
    const float* C_im   = (const float*)d_in[9];
    const float* Dp     = (const float*)d_in[10];
    const float* Wo     = (const float*)d_in[11];
    const float* bo     = (const float*)d_in[12];
    const float* ln_g   = (const float*)d_in[13];
    const float* ln_b   = (const float*)d_in[14];
    const float* fc_w   = (const float*)d_in[15];
    const float* fc_b   = (const float*)d_in[16];
    float* out = (float*)d_out;

    // workspace layout (~202 MB): r f32 | abuf bf16 | params | WoT bf16
    char* ws = (char*)d_ws;
    const size_t RB = (size_t)B_ * L_ * H_;
    float* rbuf          = (float*)ws;                          // 134 MB
    unsigned short* abuf = (unsigned short*)(ws + RB * 4);      //  67 MB
    float* par           = (float*)(ws + RB * 4 + RB * 2);
    float* lam_r = par;
    float* lam_i = par + H_ * NC_;
    float* ct_r  = par + 2 * H_ * NC_;
    float* ct_i  = par + 3 * H_ * NC_;
    unsigned short* wobT = (unsigned short*)(par + 4 * H_ * NC_);  // 128 KB

    pre_ln_kernel<<<B_ * L_, 256, 0, stream>>>(x, pre_w, pre_b, emb_g, emb_b, rbuf);

    for (int i = 0; i < NB_; ++i) {
        const size_t pHN = (size_t)i * H_ * NC_;
        disc_kernel<<<(H_ * NC_) / 256, 256, 0, stream>>>(
            log_dt + i * H_, A_re + pHN, A_im + pHN, C_re + pHN, C_im + pHN,
            lam_r, lam_i, ct_r, ct_i);
        wob_kernel<<<(H_ * H_) / 256, 256, 0, stream>>>(Wo + (size_t)i * H_ * H_, wobT);
        scan_kernel<<<(B_ * H_) / 8, 256, 0, stream>>>(
            rbuf, lam_r, lam_i, ct_r, ct_i, Dp + i * H_, abuf);
        gemm_ln_kernel<<<(B_ * L_) / 128, 256, 0, stream>>>(
            abuf, wobT, bo + i * H_, rbuf, ln_g + i * H_, ln_b + i * H_);
    }

    head_kernel<<<B_, 256, 0, stream>>>(rbuf, fc_w, fc_b, out);
}
